// DecoderGRU_32985348833417
// MI455X (gfx1250) — matrix operand order repaired and checked by reading, not compiled, not run on hardware
//
#include <hip/hip_runtime.h>
#include <hip/hip_bf16.h>
#include <math.h>

// ---------------- problem constants ----------------
constexpr int B   = 32;
constexpr int T   = 64;
constexpr int L   = 2;
constexpr int E   = 512;
constexpr int H   = 1024;
constexpr int V   = 32000;
constexpr int SOS = 1;
constexpr int G3H = 3 * H;      // 3072

// ---------------- vector types for WMMA ----------------
typedef __attribute__((ext_vector_type(8)))  float  v8f;
typedef __attribute__((ext_vector_type(8)))  __bf16 v8bf;
typedef __attribute__((ext_vector_type(16))) __bf16 v16bf;

__device__ __forceinline__ __bf16 f2bf(float f) {
    unsigned u = __builtin_bit_cast(unsigned, f);
    u += 0x7FFFu + ((u >> 16) & 1u);          // round-to-nearest-even
    unsigned short h = (unsigned short)(u >> 16);
    return __builtin_bit_cast(__bf16, h);
}

__device__ __forceinline__ v16bf cat8(v8bf lo, v8bf hi) {
    return __builtin_shufflevector(lo, hi, 0,1,2,3,4,5,6,7,8,9,10,11,12,13,14,15);
}

// A fragment from precomputed per-lane row base (elements 0..7 at +kb, 8..15 at +kb+16)
__device__ __forceinline__ v16bf load_a_at(const __bf16* abase, int kb) {
    v8bf lo = *(const v8bf*)(abase + kb);
    v8bf hi = *(const v8bf*)(abase + kb + 16);
    return cat8(lo, hi);
}

// B fragment from precomputed per-lane row base (elements 0..7 at +off, 8..15 at +off+16)
__device__ __forceinline__ v16bf load_b_at(const __bf16* wbase, int off) {
    v8bf lo = *(const v8bf*)(wbase + off);
    v8bf hi = *(const v8bf*)(wbase + off + 16);
    return cat8(lo, hi);
}

__device__ __forceinline__ v8f wmma_bf16(v16bf a, v16bf b, v8f c) {
    return __builtin_amdgcn_wmma_f32_16x16x32_bf16(
        /*neg_a=*/false, a, /*neg_b=*/false, b,
        /*c_mod=*/(short)0, c, /*reuse_a=*/false, /*reuse_b=*/false);
}

__device__ __forceinline__ float sigmoidf_(float x) { return 1.0f / (1.0f + expf(-x)); }

// ---------------- kernels ----------------

__global__ void cvt_f32_bf16_kernel(const float* __restrict__ s, __bf16* __restrict__ d, long n) {
    long i      = (long)blockIdx.x * blockDim.x + threadIdx.x;
    long stride = (long)gridDim.x * blockDim.x;
    for (; i < n; i += stride) d[i] = f2bf(s[i]);
}

// encoder_hidden [L,B,H] -> hstate fp32/bf16 parity-0 buffers ([l*2+par]*B*H layout)
__global__ void init_h_kernel(const float* __restrict__ eh, float* __restrict__ hf,
                              __bf16* __restrict__ hbf) {
    int i = blockIdx.x * blockDim.x + threadIdx.x;
    if (i >= L * B * H) return;
    int l = i / (B * H);
    int j = i % (B * H);
    float v = eh[i];
    hf [(size_t)(l * 2) * B * H + j] = v;
    hbf[(size_t)(l * 2) * B * H + j] = f2bf(v);
}

// gather embedding rows for current tokens -> bf16 x [B,E]
__global__ void embed_kernel(const float* __restrict__ emb, const int* __restrict__ tok,
                             int t, __bf16* __restrict__ xbf) {
    int i = blockIdx.x * blockDim.x + threadIdx.x;
    if (i >= B * E) return;
    int b = i / E, e = i % E;
    int tk = (t == 0) ? SOS : tok[b];
    xbf[i] = f2bf(emb[(size_t)tk * E + e]);
}

// One GRU cell step: one block (8 waves) per 16-column slice of H; K split across waves.
// Accumulators per wave: [0]=r m0, [1]=r m1, [2]=z m0, [3]=z m1,
//                        [4]=i_n m0, [5]=i_n m1, [6]=h_n m0, [7]=h_n m1.
// grid = H/16 = 64 blocks, block = 256 threads.
template <int KIN>
__global__ __launch_bounds__(256) void gru_cell_kernel(
    const __bf16* __restrict__ Xbf,   // [B, KIN]
    const __bf16* __restrict__ Hbf,   // [B, H] prev hidden (bf16)
    const float*  __restrict__ Hprev, // [B, H] prev hidden (fp32)
    const __bf16* __restrict__ Wih,   // [3H, KIN]
    const __bf16* __restrict__ Whh,   // [3H, H]
    const float*  __restrict__ bih,
    const float*  __restrict__ bhh,
    float*  __restrict__ Hout,        // [B, H]
    __bf16* __restrict__ HbfOut) {    // [B, H]
    constexpr int KSX   = KIN / 32;       // x-phase k-steps
    constexpr int KSH   = H / 32;         // h-phase k-steps
    constexpr int KS    = KSX + KSH;      // 48 (layer0) or 64 (layer1)
    constexpr int CHUNK = KS / 8;         // 6 or 8, divides evenly

    __shared__ float red[8][8][256];      // [wave][acc][lane*8+vv]  (64 KB)

    int lane = threadIdx.x & 31;
    int wave = threadIdx.x >> 5;
    int half = lane >> 4;
    int lrow = lane & 15;
    int n0   = blockIdx.x * 16;

    v8f acc[8];
#pragma unroll
    for (int a = 0; a < 8; ++a) acc[a] = (v8f)0.0f;

    int ks0   = wave * CHUNK;
    int ksend = ks0 + CHUNK;

    // ---- x phase: gi partials (k-steps [ks0, min(ksend,KSX)) )
    {
        int s1 = ksend < KSX ? ksend : KSX;
        const __bf16* ab0 = Xbf + (size_t)lrow * KIN + half * 8;
        const __bf16* ab1 = Xbf + (size_t)(16 + lrow) * KIN + half * 8;
        const __bf16* wb  = Wih + (size_t)(n0 + lrow) * KIN + half * 8;
#pragma unroll 2
        for (int ks = ks0; ks < s1; ++ks) {
            int kb = ks * 32;
            v16bf a0 = load_a_at(ab0, kb);
            v16bf a1 = load_a_at(ab1, kb);
            v16bf br = load_b_at(wb, kb);                 // gate r rows
            v16bf bz = load_b_at(wb, kb + H * KIN);       // gate z rows (const offset)
            v16bf bn = load_b_at(wb, kb + 2 * H * KIN);   // gate n rows (const offset)
            acc[0] = wmma_bf16(a0, br, acc[0]);
            acc[1] = wmma_bf16(a1, br, acc[1]);
            acc[2] = wmma_bf16(a0, bz, acc[2]);
            acc[3] = wmma_bf16(a1, bz, acc[3]);
            acc[4] = wmma_bf16(a0, bn, acc[4]);
            acc[5] = wmma_bf16(a1, bn, acc[5]);
        }
    }
    // ---- h phase: gh partials (k-steps [max(ks0,KSX)-KSX, ksend-KSX) )
    {
        int s0 = (ks0 > KSX ? ks0 : KSX) - KSX;
        int s1 = ksend - KSX;
        const __bf16* ab0 = Hbf + (size_t)lrow * H + half * 8;
        const __bf16* ab1 = Hbf + (size_t)(16 + lrow) * H + half * 8;
        const __bf16* wb  = Whh + (size_t)(n0 + lrow) * H + half * 8;
#pragma unroll 2
        for (int ks = s0; ks < s1; ++ks) {
            int kb = ks * 32;
            v16bf a0 = load_a_at(ab0, kb);
            v16bf a1 = load_a_at(ab1, kb);
            v16bf br = load_b_at(wb, kb);
            v16bf bz = load_b_at(wb, kb + H * H);
            v16bf bn = load_b_at(wb, kb + 2 * H * H);
            acc[0] = wmma_bf16(a0, br, acc[0]);
            acc[1] = wmma_bf16(a1, br, acc[1]);
            acc[2] = wmma_bf16(a0, bz, acc[2]);
            acc[3] = wmma_bf16(a1, bz, acc[3]);
            acc[6] = wmma_bf16(a0, bn, acc[6]);
            acc[7] = wmma_bf16(a1, bn, acc[7]);
        }
    }

    // stage per-wave partials to LDS
#pragma unroll
    for (int a = 0; a < 8; ++a)
#pragma unroll
        for (int vv = 0; vv < 8; ++vv)
            red[wave][a][lane * 8 + vv] = acc[a][vv];
    __syncthreads();

    // cross-wave reduction: thread e owns element e of every accumulator
    {
        int e = threadIdx.x;
#pragma unroll
        for (int a = 0; a < 8; ++a) {
            float s = 0.0f;
#pragma unroll
            for (int w = 0; w < 8; ++w) s += red[w][a][e];
            red[0][a][e] = s;
        }
    }
    __syncthreads();

    // wave 0: fused GRU nonlinearity epilogue
    if (wave == 0) {
        int col = n0 + lrow;
        float b_r  = bih[0 * H + col] + bhh[0 * H + col];
        float b_z  = bih[1 * H + col] + bhh[1 * H + col];
        float b_in = bih[2 * H + col];
        float b_hn = bhh[2 * H + col];
#pragma unroll
        for (int mt = 0; mt < 2; ++mt) {
#pragma unroll
            for (int vv = 0; vv < 8; ++vv) {
                int e = lane * 8 + vv;
                float rr = sigmoidf_(red[0][0 + mt][e] + b_r);
                float zz = sigmoidf_(red[0][2 + mt][e] + b_z);
                float nn = tanhf(red[0][4 + mt][e] + b_in + rr * (red[0][6 + mt][e] + b_hn));
                int m = mt * 16 + half * 8 + vv;          // batch row
                float hp = Hprev[(size_t)m * H + col];
                float hn = (1.0f - zz) * nn + zz * hp;
                Hout  [(size_t)m * H + col] = hn;
                HbfOut[(size_t)m * H + col] = f2bf(hn);
            }
        }
    }
}

// logits[b, t, v] = h1 @ Wout^T + b_out, written directly into d_out.
// Each wave owns LNT=4 consecutive 16-col tiles (64 rows of Wout); A regs reused 4x.
// All loads at immediate offsets from hoisted bases -> clause/pipeline friendly.
// 2 waves/block; 2000/4/2 = 250 blocks.
constexpr int LNT = 4;
__global__ __launch_bounds__(64) void logits_kernel(
    const __bf16* __restrict__ Hbf,    // [B, H]
    const __bf16* __restrict__ Wout,   // [V, H]
    const float*  __restrict__ bout,   // [V]
    float* __restrict__ out,           // [B, T, V]
    int t) {
    int lane = threadIdx.x & 31;
    int wave = threadIdx.x >> 5;
    int half = lane >> 4;
    int lrow = lane & 15;
    int wid  = blockIdx.x * 2 + wave;          // 0..499
    int n0   = wid * (LNT * 16);

    const __bf16* ab0 = Hbf + (size_t)lrow * H + half * 8;
    const __bf16* ab1 = Hbf + (size_t)(16 + lrow) * H + half * 8;
    const __bf16* wb  = Wout + (size_t)(n0 + lrow) * H + half * 8;

    v8f acc[LNT][2];
#pragma unroll
    for (int j = 0; j < LNT; ++j) { acc[j][0] = (v8f)0.0f; acc[j][1] = (v8f)0.0f; }

#pragma unroll 2
    for (int kb = 0; kb < H; kb += 32) {
        v16bf a0 = load_a_at(ab0, kb);
        v16bf a1 = load_a_at(ab1, kb);
        v16bf b0 = load_b_at(wb, kb);                 // tile j at const offset j*16*H
        v16bf b1 = load_b_at(wb, kb + 16 * H);
        v16bf b2 = load_b_at(wb, kb + 32 * H);
        v16bf b3 = load_b_at(wb, kb + 48 * H);
        acc[0][0] = wmma_bf16(a0, b0, acc[0][0]);
        acc[0][1] = wmma_bf16(a1, b0, acc[0][1]);
        acc[1][0] = wmma_bf16(a0, b1, acc[1][0]);
        acc[1][1] = wmma_bf16(a1, b1, acc[1][1]);
        acc[2][0] = wmma_bf16(a0, b2, acc[2][0]);
        acc[2][1] = wmma_bf16(a1, b2, acc[2][1]);
        acc[3][0] = wmma_bf16(a0, b3, acc[3][0]);
        acc[3][1] = wmma_bf16(a1, b3, acc[3][1]);
    }

#pragma unroll
    for (int j = 0; j < LNT; ++j) {
        int col  = n0 + j * 16 + lrow;
        float bo = bout[col];
#pragma unroll
        for (int vv = 0; vv < 8; ++vv) {
            int m0 = half * 8 + vv;
            int m1 = 16 + half * 8 + vv;
            out[(size_t)m0 * T * V + (size_t)t * V + col] = acc[j][0][vv] + bo;
            out[(size_t)m1 * T * V + (size_t)t * V + col] = acc[j][1][vv] + bo;
        }
    }
}

// Per batch row: argmax (-> next token) + in-place log_softmax over V.
__global__ void softmax_argmax_kernel(float* __restrict__ out, int t, int* __restrict__ tok) {
    int b   = blockIdx.x;
    int tid = threadIdx.x;
    float* row = out + (size_t)b * T * V + (size_t)t * V;

    __shared__ float sv[256];
    __shared__ int   si[256];

    float bestv = -INFINITY;
    int   besti = 0;
    for (int i = tid; i < V; i += 256) {
        float v = row[i];
        if (v > bestv || (v == bestv && i < besti)) { bestv = v; besti = i; }
    }
    sv[tid] = bestv; si[tid] = besti;
    __syncthreads();
    for (int s = 128; s > 0; s >>= 1) {
        if (tid < s) {
            float ov = sv[tid + s]; int oi = si[tid + s];
            if (ov > sv[tid] || (ov == sv[tid] && oi < si[tid])) { sv[tid] = ov; si[tid] = oi; }
        }
        __syncthreads();
    }
    float rmax = sv[0];
    if (tid == 0) tok[b] = si[0];
    __syncthreads();

    float lsum = 0.0f;
    for (int i = tid; i < V; i += 256) lsum += expf(row[i] - rmax);
    sv[tid] = lsum;
    __syncthreads();
    for (int s = 128; s > 0; s >>= 1) {
        if (tid < s) sv[tid] += sv[tid + s];
        __syncthreads();
    }
    float lse = rmax + logf(sv[0]);
    __syncthreads();
    for (int i = tid; i < V; i += 256) row[i] = row[i] - lse;
}

// copy final hidden state (parity 0 after 64 steps) to d_out tail, layout [L,B,H]
__global__ void final_h_kernel(const float* __restrict__ hf, float* __restrict__ outTail) {
    int i = blockIdx.x * blockDim.x + threadIdx.x;
    if (i >= L * B * H) return;
    int l = i / (B * H);
    int j = i % (B * H);
    outTail[i] = hf[(size_t)(l * 2 + 0) * B * H + j];
}

// ---------------- host launcher ----------------
extern "C" void kernel_launch(void* const* d_in, const int* in_sizes, int n_in,
                              void* d_out, int out_size, void* d_ws, size_t ws_size,
                              hipStream_t stream) {
    (void)in_sizes; (void)n_in; (void)out_size; (void)ws_size;
    const float* enc_out = (const float*)d_in[0];  (void)enc_out; // unused by reference
    const float* enc_hid = (const float*)d_in[1];
    const float* emb     = (const float*)d_in[2];
    const float* w_ih0   = (const float*)d_in[3];
    const float* w_hh0   = (const float*)d_in[4];
    const float* b_ih0   = (const float*)d_in[5];
    const float* b_hh0   = (const float*)d_in[6];
    const float* w_ih1   = (const float*)d_in[7];
    const float* w_hh1   = (const float*)d_in[8];
    const float* b_ih1   = (const float*)d_in[9];
    const float* b_hh1   = (const float*)d_in[10];
    const float* w_out   = (const float*)d_in[11];
    const float* b_out   = (const float*)d_in[12];
    float* out = (float*)d_out;

    // carve workspace (256B aligned slices)
    char*  ws  = (char*)d_ws;
    size_t off = 0;
    auto carve = [&](size_t bytes) -> void* {
        off = (off + 255) & ~(size_t)255;
        void* p = ws + off;
        off += bytes;
        return p;
    };
    __bf16* wih0_bf = (__bf16*)carve((size_t)G3H * E * 2);
    __bf16* whh0_bf = (__bf16*)carve((size_t)G3H * H * 2);
    __bf16* wih1_bf = (__bf16*)carve((size_t)G3H * H * 2);
    __bf16* whh1_bf = (__bf16*)carve((size_t)G3H * H * 2);
    __bf16* wout_bf = (__bf16*)carve((size_t)V * H * 2);
    float*  hf      = (float*) carve((size_t)L * 2 * B * H * 4);   // [l][parity][B][H]
    __bf16* hbf     = (__bf16*)carve((size_t)L * 2 * B * H * 2);
    __bf16* xbf     = (__bf16*)carve((size_t)B * E * 2);
    int*    tokbuf  = (int*)   carve((size_t)B * 4);

    auto cvt = [&](const float* s, __bf16* d, long n) {
        int blocks = (int)((n + 255) / 256);
        if (blocks > 2048) blocks = 2048;
        cvt_f32_bf16_kernel<<<blocks, 256, 0, stream>>>(s, d, n);
    };
    cvt(w_ih0, wih0_bf, (long)G3H * E);
    cvt(w_hh0, whh0_bf, (long)G3H * H);
    cvt(w_ih1, wih1_bf, (long)G3H * H);
    cvt(w_hh1, whh1_bf, (long)G3H * H);
    cvt(w_out, wout_bf, (long)V * H);

    init_h_kernel<<<(L * B * H + 255) / 256, 256, 0, stream>>>(enc_hid, hf, hbf);

    const size_t BH = (size_t)B * H;
    for (int t = 0; t < T; ++t) {
        int pin  = t & 1;
        int pout = (t + 1) & 1;

        embed_kernel<<<(B * E + 255) / 256, 256, 0, stream>>>(emb, tokbuf, t, xbf);

        // layer 0: x [B,E]
        gru_cell_kernel<E><<<H / 16, 256, 0, stream>>>(
            xbf,
            hbf + (0 * 2 + pin) * BH, hf + (0 * 2 + pin) * BH,
            wih0_bf, whh0_bf, b_ih0, b_hh0,
            hf + (0 * 2 + pout) * BH, hbf + (0 * 2 + pout) * BH);

        // layer 1: x = h0_new [B,H]
        gru_cell_kernel<H><<<H / 16, 256, 0, stream>>>(
            hbf + (0 * 2 + pout) * BH,
            hbf + (1 * 2 + pin) * BH, hf + (1 * 2 + pin) * BH,
            wih1_bf, whh1_bf, b_ih1, b_hh1,
            hf + (1 * 2 + pout) * BH, hbf + (1 * 2 + pout) * BH);

        // output head: [B,V] logits -> d_out[:, t, :]
        logits_kernel<<<V / 16 / LNT / 2, 64, 0, stream>>>(
            hbf + (1 * 2 + pout) * BH, wout_bf, b_out, out, t);

        // argmax -> next token; in-place log_softmax
        softmax_argmax_kernel<<<B, 256, 0, stream>>>(out, t, tokbuf);
    }

    final_h_kernel<<<(L * B * H + 255) / 256, 256, 0, stream>>>(
        hf, out + (size_t)B * T * V);
}
